// BahdanauAttention_45698452029995
// MI455X (gfx1250) — matrix operand order repaired and checked by reading, not compiled, not run on hardware
//
#include <hip/hip_runtime.h>
#include <hip/hip_bf16.h>
#include <math.h>

typedef __attribute__((ext_vector_type(16))) __bf16 v16bf;
typedef __attribute__((ext_vector_type(8)))  float  v8f;

#define B_DIM 32
#define T_DIM 4096
#define D_DIM 512

__device__ __forceinline__ float fast_tanh(float x) {
    // tanh(x) = 1 - 2/(exp(2x)+1); exact at +/-inf, ~1 ulp elsewhere
    float e = __expf(2.0f * x);
    return 1.0f - 2.0f * __builtin_amdgcn_rcpf(e + 1.0f);
}

// ---------------------------------------------------------------------------
// Prep 1: pack W_h (f32, [k][n] row-major 512x512) into bf16 WMMA-B-fragment
// order. B matrix is 32x16 (KxN) per 16x16x32 step: lane holds n = 16*nt+l%16,
// lanes 0-15 K=kc*32+0..15, lanes 16-31 K=kc*32+16..31 (contiguous 32B/lane).
// Layout: chunk (nt, kc, lane) -> 16 bf16 at [((nt*16+kc)*32+lane)*16].
// ---------------------------------------------------------------------------
__global__ void pack_wh_kernel(const float* __restrict__ Wh, __bf16* __restrict__ out) {
    int g = blockIdx.x * 256 + threadIdx.x;      // 262144 elements
    int e     = g & 15;
    int chunk = g >> 4;
    int lane  = chunk & 31;
    int kc    = (chunk >> 5) & 15;
    int nt    = chunk >> 9;
    int n = nt * 16 + (lane & 15);
    int k = kc * 32 + (lane >> 4) * 16 + e;
    out[g] = (__bf16)Wh[k * D_DIM + n];
}

// ---------------------------------------------------------------------------
// Prep 2: dec_features[b][n] = s_t[b] @ W_s[:,n] + b_s[n]   (32x512, tiny)
// ---------------------------------------------------------------------------
__global__ void dec_feat_kernel(const float* __restrict__ s_t, const float* __restrict__ Ws,
                                const float* __restrict__ bs, float* __restrict__ dec) {
    int g = blockIdx.x * 256 + threadIdx.x;      // 16384
    int b = g >> 9, n = g & 511;
    float acc = bs[n];
    for (int k = 0; k < D_DIM; ++k)
        acc += s_t[b * D_DIM + k] * Ws[k * D_DIM + n];
    dec[g] = acc;
}

// ---------------------------------------------------------------------------
// Main: e[b][t] = sum_n V[n]*tanh( (h@Wh)[row][n] + dec[b][n] + cov[b][t]*Wc[n] )
// Block = 128 threads (4 waves), 128 rows. A panel staged f32->bf16 into LDS
// in A-fragment layout [kc][ms][c][lane] (8KB window per kc -> DS immediates).
// Each wave owns 8 N-tiles, all 8 M-subtiles.
// ---------------------------------------------------------------------------
__global__ __launch_bounds__(128)
__attribute__((amdgpu_waves_per_eu(2)))
void energy_kernel(const float* __restrict__ h, const float* __restrict__ cov,
                   const __bf16* __restrict__ Wpk, const float* __restrict__ dec,
                   const float* __restrict__ Wc, const float* __restrict__ V,
                   float* __restrict__ e_out) {
    __shared__ uint4 Apanel[8192];   // 128KB: [kc][ms][c][lane] 16B chunks
    __shared__ float e_sm[128];
    __shared__ float cov_sm[128];

    const int tid  = threadIdx.x;
    const int lane = tid & 31;
    const int wid  = tid >> 5;
    const int row0 = blockIdx.x * 128;
    const int b    = row0 >> 12;            // T = 4096
    const int t0   = row0 & (T_DIM - 1);

    e_sm[tid] = 0.0f;
    cov_sm[tid] = cov[b * T_DIM + t0 + tid];

    // ---- stage A: 128 rows x 512 K, f32 -> bf16, into fragment layout ----
    const float* hbase = h + (size_t)row0 * D_DIM;
    for (int it = 0; it < 64; ++it) {
        int task = it * 128 + tid;
        int R = task >> 6;                   // row 0..127 (coalesced groups)
        int g = task & 63;                   // 8-float group along K
        const float4* p = (const float4*)(hbase + (size_t)R * D_DIM + g * 8);
        float4 f0 = p[0], f1 = p[1];
        union { __bf16 h8[8]; uint4 q; } u;
        u.h8[0] = (__bf16)f0.x; u.h8[1] = (__bf16)f0.y;
        u.h8[2] = (__bf16)f0.z; u.h8[3] = (__bf16)f0.w;
        u.h8[4] = (__bf16)f1.x; u.h8[5] = (__bf16)f1.y;
        u.h8[6] = (__bf16)f1.z; u.h8[7] = (__bf16)f1.w;
        int kc  = g >> 2;
        int rem = g & 3;                     // K sub-offset 8*rem in the 32-chunk
        int hi  = rem >> 1;                  // lane half
        int c   = rem & 1;                   // fragment chunk 0/1
        int l   = hi * 16 + (R & 15);
        int ms  = R >> 4;
        Apanel[((kc * 8 + ms) * 2 + c) * 32 + l] = u.q;
    }
    __syncthreads();

    const int nhalf = lane >> 4;
    const int nlo   = lane & 15;

    float e_part[8][8];
#pragma unroll
    for (int ms = 0; ms < 8; ++ms)
#pragma unroll
        for (int r = 0; r < 8; ++r) e_part[ms][r] = 0.0f;

    for (int j = 0; j < 8; ++j) {
        const int nt = wid * 8 + j;
        const int n  = nt * 16 + nlo;
        const float dec_v = dec[b * D_DIM + n];
        const float wc    = Wc[n];
        const float vv    = V[n];

        v8f acc[8];
#pragma unroll
        for (int ms = 0; ms < 8; ++ms)
            acc[ms] = (v8f){0.f, 0.f, 0.f, 0.f, 0.f, 0.f, 0.f, 0.f};

        // per-j base pointers: all kc/ms accesses become immediate offsets
        const uint4* bp = (const uint4*)Wpk + (size_t)(nt * 16 * 32 + lane) * 2;

#pragma unroll 2
        for (int kc = 0; kc < 16; ++kc) {
            union { uint4 q[2]; v16bf v; } bu;
            bu.q[0] = bp[kc * 64];           // +1024B per kc
            bu.q[1] = bp[kc * 64 + 1];
            const uint4* ap = Apanel + kc * 512 + lane;  // 8KB window per kc
#pragma unroll
            for (int ms = 0; ms < 8; ++ms) {
                union { uint4 q[2]; v16bf v; } au;
                au.q[0] = ap[ms * 64];       // c=0
                au.q[1] = ap[ms * 64 + 32];  // c=1 (+512B)
                acc[ms] = __builtin_amdgcn_wmma_f32_16x16x32_bf16(
                    false, au.v, false, bu.v, (short)0, acc[ms], false, false);
            }
        }
        // fused epilogue: tanh + dot-with-V partials (per-lane, reduced later)
#pragma unroll
        for (int ms = 0; ms < 8; ++ms) {
#pragma unroll
            for (int r = 0; r < 8; ++r) {
                float cv = cov_sm[ms * 16 + r + nhalf * 8];
                float f  = acc[ms][r] + dec_v + cv * wc;
                e_part[ms][r] += vv * fast_tanh(f);
            }
        }
    }

    // reduce over the 16 lanes that hold one row's N-slice, then combine waves
#pragma unroll
    for (int ms = 0; ms < 8; ++ms) {
#pragma unroll
        for (int r = 0; r < 8; ++r) {
            float v = e_part[ms][r];
            v += __shfl_xor(v, 1, 32);
            v += __shfl_xor(v, 2, 32);
            v += __shfl_xor(v, 4, 32);
            v += __shfl_xor(v, 8, 32);
            if (nlo == 0) atomicAdd(&e_sm[ms * 16 + r + nhalf * 8], v);
        }
    }
    __syncthreads();
    e_out[b * T_DIM + t0 + tid] = e_sm[tid];
}

// ---------------------------------------------------------------------------
// Softmax over T per batch; writes a_t and new_coverage.
// ---------------------------------------------------------------------------
__global__ __launch_bounds__(256)
void softmax_kernel(const float* __restrict__ e, const float* __restrict__ cov,
                    float* __restrict__ a_out, float* __restrict__ ncov_out) {
    __shared__ float buf[T_DIM];
    __shared__ float red[256];
    const int b = blockIdx.x, tid = threadIdx.x;
    const float* er = e + (size_t)b * T_DIM;

    float mx = -1e30f;
    for (int i = tid; i < T_DIM; i += 256) {
        float v = er[i];
        buf[i] = v;
        mx = fmaxf(mx, v);
    }
    red[tid] = mx;
    __syncthreads();
    for (int s = 128; s > 0; s >>= 1) {
        if (tid < s) red[tid] = fmaxf(red[tid], red[tid + s]);
        __syncthreads();
    }
    mx = red[0];
    __syncthreads();

    float sum = 0.0f;
    for (int i = tid; i < T_DIM; i += 256) {
        float ev = __expf(buf[i] - mx);
        buf[i] = ev;
        sum += ev;
    }
    red[tid] = sum;
    __syncthreads();
    for (int s = 128; s > 0; s >>= 1) {
        if (tid < s) red[tid] += red[tid + s];
        __syncthreads();
    }
    const float inv = 1.0f / red[0];
    for (int i = tid; i < T_DIM; i += 256) {
        float a = buf[i] * inv;
        a_out[(size_t)b * T_DIM + i]    = a;
        ncov_out[(size_t)b * T_DIM + i] = cov[(size_t)b * T_DIM + i] + a;
    }
}

// ---------------------------------------------------------------------------
// Context: ctx[b][d] = sum_t a[b][t] * h[b][t][d]  (2nd pass over h_i)
// ---------------------------------------------------------------------------
__global__ void zero_ctx_kernel(float* __restrict__ ctx) {
    int g = blockIdx.x * 256 + threadIdx.x;
    if (g < B_DIM * D_DIM) ctx[g] = 0.0f;
}

__global__ __launch_bounds__(256)
void context_kernel(const float* __restrict__ h, const float* __restrict__ a,
                    float* __restrict__ ctx) {
    const int blk = blockIdx.x;              // 512 = 32 b * 16 t-chunks
    const int b = blk >> 4, tc = blk & 15;
    const int tid = threadIdx.x;
    float acc0 = 0.0f, acc1 = 0.0f;
    const size_t rbase = (size_t)b * T_DIM + tc * 256;
    for (int i = 0; i < 256; ++i) {
        float av = a[rbase + i];
        const float* hp = h + (rbase + i) * D_DIM;
        acc0 += av * hp[tid];
        acc1 += av * hp[tid + 256];
    }
    atomicAdd(&ctx[b * D_DIM + tid], acc0);
    atomicAdd(&ctx[b * D_DIM + tid + 256], acc1);
}

// ---------------------------------------------------------------------------
extern "C" void kernel_launch(void* const* d_in, const int* in_sizes, int n_in,
                              void* d_out, int out_size, void* d_ws, size_t ws_size,
                              hipStream_t stream) {
    const float* h_i  = (const float*)d_in[0];
    const float* s_t  = (const float*)d_in[1];
    const float* covp = (const float*)d_in[2];
    const float* W_h  = (const float*)d_in[3];
    const float* W_s  = (const float*)d_in[4];
    const float* b_s  = (const float*)d_in[5];
    const float* W_c  = (const float*)d_in[6];
    const float* V    = (const float*)d_in[7];

    // workspace: Wpk bf16 512KB | dec 64KB | e 512KB
    char* ws = (char*)d_ws;
    __bf16* Wpk = (__bf16*)ws;
    float*  dec = (float*)(ws + (512 << 10));
    float*  e   = (float*)(ws + (576 << 10));

    float* out  = (float*)d_out;
    float* ctx  = out;                                   // (B, D)
    float* a_t  = out + B_DIM * D_DIM;                   // (B, T)
    float* ncov = out + B_DIM * D_DIM + B_DIM * T_DIM;   // (B, T)

    pack_wh_kernel<<<1024, 256, 0, stream>>>(W_h, Wpk);
    dec_feat_kernel<<<64, 256, 0, stream>>>(s_t, W_s, b_s, dec);
    energy_kernel<<<(B_DIM * T_DIM) / 128, 128, 0, stream>>>(h_i, covp, Wpk, dec, W_c, V, e);
    softmax_kernel<<<B_DIM, 256, 0, stream>>>(e, covp, a_t, ncov);
    zero_ctx_kernel<<<(B_DIM * D_DIM + 255) / 256, 256, 0, stream>>>(ctx);
    context_kernel<<<B_DIM * 16, 256, 0, stream>>>(h_i, a_t, ctx);
}
